// ScaledDotProductAttention_21775484191276
// MI455X (gfx1250) — matrix operand order repaired and checked by reading, not compiled, not run on hardware
//
#include <hip/hip_runtime.h>

// ---- CDNA5 (gfx1250) wave32 WMMA flash-attention forward -------------------
// D = A(16x32 f16) * B(32x16 f16) + C(16x16 f32) via v_wmma_f32_16x16x32_f16.
// 64-key blocks, 8 waves/WG (128 q-rows), DPP16 softmax reductions,
// double-buffered LDS staging with one barrier per key block.

typedef _Float16 v16h __attribute__((ext_vector_type(16)));
typedef float    v8f  __attribute__((ext_vector_type(8)));
typedef _Float16 h2   __attribute__((ext_vector_type(2)));
typedef float    f2   __attribute__((ext_vector_type(2)));
typedef float    f4   __attribute__((ext_vector_type(4)));

#define WMMA_F16(a, b, c) \
  __builtin_amdgcn_wmma_f32_16x16x32_f16(false, (a), false, (b), (short)0, (c), false, false)

namespace {
constexpr int   kS      = 2048;
constexpr int   kD      = 64;
constexpr float kScale  = 0.125f;
constexpr int   kWaves  = 8;               // waves per workgroup
constexpr int   kThreads= kWaves * 32;     // 256
constexpr int   kQRows  = kWaves * 16;     // 128 query rows per workgroup
constexpr int   kKBlk   = 64;              // key block (2 x WMMA f16 K)
constexpr int   kKStr   = 72;              // LDS halves/row for K  [key][d], 64 + 8 pad
constexpr int   kVStr   = 72;              // LDS halves/row for V^T [d][key], 64 + 8 pad
constexpr int   kPStr   = 68;              // LDS halves/row for P tiles (64 + 4 pad)
constexpr int   kKBufH  = kKBlk * kKStr;   // halves per K buffer
constexpr int   kVBufH  = kD * kVStr;      // halves per V buffer
}

// 16-lane butterfly reduction via DPP16 (pure VALU, no LDS / dscnt waits).
__device__ __forceinline__ float dpp_xchg(float x, int ctrl_tag) {
  const int xi = __builtin_bit_cast(int, x);
  int r;
  switch (ctrl_tag) {
    case 0:  r = __builtin_amdgcn_update_dpp(0, xi, 0xB1,  0xF, 0xF, true); break; // quad_perm(1,0,3,2)
    case 1:  r = __builtin_amdgcn_update_dpp(0, xi, 0x4E,  0xF, 0xF, true); break; // quad_perm(2,3,0,1)
    case 2:  r = __builtin_amdgcn_update_dpp(0, xi, 0x141, 0xF, 0xF, true); break; // row_half_mirror
    default: r = __builtin_amdgcn_update_dpp(0, xi, 0x140, 0xF, 0xF, true); break; // row_mirror
  }
  return __builtin_bit_cast(float, r);
}
__device__ __forceinline__ float row_max16(float x) {
  x = fmaxf(x, dpp_xchg(x, 0));
  x = fmaxf(x, dpp_xchg(x, 1));
  x = fmaxf(x, dpp_xchg(x, 2));
  x = fmaxf(x, dpp_xchg(x, 3));
  return x;
}
__device__ __forceinline__ float row_sum16(float x) {
  x += dpp_xchg(x, 0);
  x += dpp_xchg(x, 1);
  x += dpp_xchg(x, 2);
  x += dpp_xchg(x, 3);
  return x;
}

__global__ __launch_bounds__(kThreads)
void fa_fwd_f1250(const float* __restrict__ Q, const float* __restrict__ K,
                  const float* __restrict__ V, float* __restrict__ O) {
  __shared__ _Float16 Kl[2 * kKBufH];            // K block, [key][d], f16 (x2 buf)
  __shared__ _Float16 Vl[2 * kVBufH];            // V^T block, [d][key], f16 (x2 buf)
  __shared__ _Float16 Pl[kWaves * 16 * kPStr];   // per-wave P tile (C->A relayout)

  const int nQblk = kS / kQRows;
  const int bh    = blockIdx.x / nQblk;
  const int q0    = (blockIdx.x % nQblk) * kQRows;

  const int tid   = threadIdx.x;
  const int wave  = tid >> 5;
  const int lane  = tid & 31;
  const int lmod  = lane & 15;
  const int lhalf = lane >> 4;
  const int qw    = q0 + wave * 16;              // first query row of this wave

  const float* Qb = Q + (size_t)bh * kS * kD;
  const float* Kb = K + (size_t)bh * kS * kD;
  const float* Vb = V + (size_t)bh * kS * kD;
  float*       Ob = O + (size_t)bh * kS * kD;

  // Cooperative staging: 64x64 f32 block = 1024 float4; 4 per thread.
  const int srow[4] = { (tid + 0*kThreads) >> 4, (tid + 1*kThreads) >> 4,
                        (tid + 2*kThreads) >> 4, (tid + 3*kThreads) >> 4 };
  const int sc4     = tid & 15;

  // ---- Q A-fragments (two 16x32 f16 tiles covering d=0..63), scale folded in.
  // A layout (ISA 7.12.2, 16-bit 16x32): lane = M%16 (half = K group),
  // VGPR v holds K = (v<4?0:16) + half*8 + (v&3)*2 (+1 in high half of dword).
  v16h aq[2];
#pragma unroll
  for (int g = 0; g < 2; ++g) {
#pragma unroll
    for (int v = 0; v < 8; ++v) {
      const int d = g * 32 + ((v < 4) ? 0 : 16) + lhalf * 8 + (v & 3) * 2;
      const f2 q = *(const f2*)(Qb + (size_t)(qw + lmod) * kD + d);
      aq[g][2 * v]     = (_Float16)(q.x * kScale);
      aq[g][2 * v + 1] = (_Float16)(q.y * kScale);
    }
  }

  v8f o[4] = {};
  float m8[8], l8[8];
#pragma unroll
  for (int v = 0; v < 8; ++v) { m8[v] = -__builtin_inff(); l8[v] = 0.0f; }

  const int kend = q0 + kQRows;                  // causal: keys needed by block

  // ---- prologue: load key block 0 into registers ----
  f4 kreg[4], vreg[4];
#pragma unroll
  for (int i = 0; i < 4; ++i) {
    kreg[i] = *(const f4*)(Kb + srow[i] * kD + sc4 * 4);
    vreg[i] = *(const f4*)(Vb + srow[i] * kD + sc4 * 4);
  }

  int ib = 0;
  for (int kb = 0; kb < kend; kb += kKBlk, ib ^= 1) {
    _Float16* Kbuf = Kl + ib * kKBufH;
    _Float16* Vbuf = Vl + ib * kVBufH;

    // ---- commit staged registers (block kb) to LDS buffer ib as f16 ----
#pragma unroll
    for (int i = 0; i < 4; ++i) {
#pragma unroll
      for (int j = 0; j < 4; ++j) {
        Kbuf[srow[i] * kKStr + sc4 * 4 + j]   = (_Float16)kreg[i][j];
        Vbuf[(sc4 * 4 + j) * kVStr + srow[i]] = (_Float16)vreg[i][j];  // transpose
      }
    }
    __syncthreads();

    // ---- issue global loads for block kb+64 (latency hides behind compute) ----
    if (kb + kKBlk < kend) {
      const float* Ks = Kb + (size_t)(kb + kKBlk) * kD;
      const float* Vs = Vb + (size_t)(kb + kKBlk) * kD;
#pragma unroll
      for (int i = 0; i < 4; ++i) {
        kreg[i] = *(const f4*)(Ks + srow[i] * kD + sc4 * 4);
        vreg[i] = *(const f4*)(Vs + srow[i] * kD + sc4 * 4);
      }
    }

    if (kb <= qw + 15) {                         // block not fully masked for wave
      // ---- S[16x64] = Q * K^T : four 16x16 C tiles, 2 WMMAs each ----
      v8f s[4];
#pragma unroll
      for (int c = 0; c < 4; ++c) {
        // B layout (32x16 f16): lane = N%16 (half = K group),
        // VGPR v holds K = (v<4?0:16) + half*8 + (v&3)*2 (+1 in high half of dword).
        v16h bk0, bk1;
        const int n = c * 16 + lmod;
#pragma unroll
        for (int v = 0; v < 8; ++v) {
          const int d0 = ((v < 4) ? 0 : 16) + lhalf * 8 + (v & 3) * 2;
          const h2 p0 = *(const h2*)&Kbuf[n * kKStr + d0];
          const h2 p1 = *(const h2*)&Kbuf[n * kKStr + 32 + d0];
          bk0[2 * v] = p0.x; bk0[2 * v + 1] = p0.y;
          bk1[2 * v] = p1.x; bk1[2 * v + 1] = p1.y;
        }
        v8f acc = {};
        acc = WMMA_F16(aq[0], bk0, acc);
        acc = WMMA_F16(aq[1], bk1, acc);
        s[c] = acc;
      }

      // ---- causal mask on diagonal blocks ----
      if (kb + kKBlk - 1 > qw) {
#pragma unroll
        for (int c = 0; c < 4; ++c) {
#pragma unroll
          for (int v = 0; v < 8; ++v) {
            const int row = qw + v + 8 * lhalf;
            const int col = kb + c * 16 + lmod;
            if (col > row) s[c][v] = -__builtin_inff();
          }
        }
      }

      // ---- online softmax (C layout: lane=N, VGPR v -> row v+8*half) ----
#pragma unroll
      for (int v = 0; v < 8; ++v) {
        const float ml = row_max16(fmaxf(fmaxf(s[0][v], s[1][v]),
                                         fmaxf(s[2][v], s[3][v])));
        const float mn = fmaxf(m8[v], ml);
        const float al = __expf(m8[v] - mn);
        m8[v] = mn;
        const float p0 = __expf(s[0][v] - mn);
        const float p1 = __expf(s[1][v] - mn);
        const float p2 = __expf(s[2][v] - mn);
        const float p3 = __expf(s[3][v] - mn);
        const float rs = row_sum16((p0 + p1) + (p2 + p3));
        l8[v] = l8[v] * al + rs;
        o[0][v] *= al; o[1][v] *= al; o[2][v] *= al; o[3][v] *= al;
        // spill P (C layout) row-major to wave-private LDS tile
        const int prow = (wave * 16 + v + 8 * lhalf) * kPStr;
        Pl[prow +  0 + lmod] = (_Float16)p0;
        Pl[prow + 16 + lmod] = (_Float16)p1;
        Pl[prow + 32 + lmod] = (_Float16)p2;
        Pl[prow + 48 + lmod] = (_Float16)p3;
      }

      // ---- O[16x64] += P(16x64) * V(64x64) : 2 kk-halves x 4 d-frags ----
#pragma unroll
      for (int g = 0; g < 2; ++g) {
        // regather P half g as A fragment (same-wave LDS ops are in-order)
        v16h ap;
#pragma unroll
        for (int v = 0; v < 8; ++v) {
          const int k = g * 32 + ((v < 4) ? 0 : 16) + lhalf * 8 + (v & 3) * 2;
          const h2 pp = *(const h2*)&Pl[(wave * 16 + lmod) * kPStr + k];
          ap[2 * v] = pp.x; ap[2 * v + 1] = pp.y;
        }
#pragma unroll
        for (int f = 0; f < 4; ++f) {
          v16h bv;
          const int d = f * 16 + lmod;
#pragma unroll
          for (int v = 0; v < 8; ++v) {
            const int kk = g * 32 + ((v < 4) ? 0 : 16) + lhalf * 8 + (v & 3) * 2;
            const h2 vv = *(const h2*)&Vbuf[d * kVStr + kk];
            bv[2 * v] = vv.x; bv[2 * v + 1] = vv.y;
          }
          o[f] = WMMA_F16(ap, bv, o[f]);
        }
      }
    }
    // single barrier per iteration: buffer ib is reused two iterations from
    // now; every wave's next-iteration barrier orders those writes after all
    // of this iteration's reads.
  }

  // ---- epilogue: normalize rows and store (coalesced across lmod) ----
#pragma unroll
  for (int v = 0; v < 8; ++v) {
    const float inv = 1.0f / l8[v];
    const int row = qw + v + 8 * lhalf;
#pragma unroll
    for (int f = 0; f < 4; ++f)
      Ob[(size_t)row * kD + f * 16 + lmod] = o[f][v] * inv;
  }
}

extern "C" void kernel_launch(void* const* d_in, const int* in_sizes, int n_in,
                              void* d_out, int out_size, void* d_ws, size_t ws_size,
                              hipStream_t stream) {
  (void)in_sizes; (void)n_in; (void)out_size; (void)d_ws; (void)ws_size;
  const float* Q = (const float*)d_in[0];
  const float* K = (const float*)d_in[1];
  const float* V = (const float*)d_in[2];
  float*       O = (float*)d_out;

  const int B = 4, H = 16;
  const int nblocks = B * H * (kS / kQRows);  // 1024
  fa_fwd_f1250<<<nblocks, kThreads, 0, stream>>>(Q, K, V, O);
}
